// Decoder_15951508538119
// MI455X (gfx1250) — matrix operand order repaired and checked by reading, not compiled, not run on hardware
//
#include <hip/hip_runtime.h>
#include <hip/hip_bf16.h>

#define NB 8
#define EE 512
#define TT 2048
#define HH 8
#define DD 64
#define E3 1536

typedef __attribute__((ext_vector_type(16))) __bf16 v16bf;
typedef __attribute__((ext_vector_type(8)))  __bf16 v8bf;
typedef __attribute__((ext_vector_type(8)))  float  v8f;
typedef __attribute__((ext_vector_type(8)))  unsigned short us8;

__device__ __forceinline__ unsigned short f2bf(float f) {
  unsigned u = __builtin_bit_cast(unsigned, f);
  u += 0x7FFFu + ((u >> 16) & 1u);          // round-to-nearest-even
  return (unsigned short)(u >> 16);
}

__device__ __forceinline__ v8f wmma_bf16(v16bf a, v16bf b, v8f c) {
  return __builtin_amdgcn_wmma_f32_16x16x32_bf16(false, a, false, b, (short)0, c, false, false);
}

// A fragment (16x32 bf16) from LDS tile stored [m][k], row stride ld (mult of 8), k0 mult of 32.
__device__ __forceinline__ v16bf fragA(const unsigned short* L, int ld, int mbase, int k0, int lane) {
  int m  = mbase + (lane & 15);
  int ko = (lane & 16) ? 8 : 0;
  const unsigned short* p = L + m * ld + k0 + ko;
  v8bf lo = *(const v8bf*)(p);
  v8bf hi = *(const v8bf*)(p + 16);
  return __builtin_shufflevector(lo, hi, 0,1,2,3,4,5,6,7,8,9,10,11,12,13,14,15);
}

// B fragment (32x16 bf16) from LDS stored TRANSPOSED: BT[n][k] -> two 8-ushort runs per lane (k0+8h, k0+16+8h).
__device__ __forceinline__ v16bf fragBT(const unsigned short* L, int ld, int nbase, int k0, int lane) {
  int n  = nbase + (lane & 15);
  int ko = (lane & 16) ? 8 : 0;
  const unsigned short* p = L + n * ld + k0 + ko;
  v8bf lo = *(const v8bf*)(p);
  v8bf hi = *(const v8bf*)(p + 16);
  return __builtin_shufflevector(lo, hi, 0,1,2,3,4,5,6,7,8,9,10,11,12,13,14,15);
}

// ---------------- kernel 1: weights -> bf16.  w_in/w_out already [n][k]; w_ct transposed.
__global__ void convert_weights(const float* __restrict__ w_in, const float* __restrict__ w_out,
                                const float* __restrict__ w_ct,
                                unsigned short* __restrict__ w_inB,
                                unsigned short* __restrict__ w_outB,
                                unsigned short* __restrict__ w_ctB) {
  int idx = blockIdx.x * blockDim.x + threadIdx.x;
  int stride = gridDim.x * blockDim.x;
  for (int i = idx; i < E3 * EE; i += stride) w_inB[i] = f2bf(w_in[i]);
  for (int i = idx; i < EE * EE; i += stride) {
    w_outB[i] = f2bf(w_out[i]);
    int o = i / EE, k = i % EE;
    w_ctB[i] = f2bf(w_ct[(size_t)k * EE + o]);
  }
}

// ---------------- kernel 2: QKV = mask^T @ w_in^T + b_in (q scaled by 1/8) ----------------
// 128x64 block tile, 8 waves (4m x 2n), 32x32 per wave, double-buffered LDS.
__global__ __launch_bounds__(256) void qkv_kernel(const float* __restrict__ mask,
                                                  const unsigned short* __restrict__ w_inB,
                                                  const float* __restrict__ b_in,
                                                  unsigned short* __restrict__ qkv) {
  __shared__ unsigned short As[2][128 * 40];   // [m=t][k]
  __shared__ unsigned short Bs[2][64 * 40];    // BT: [n=c][k]
  int tid = threadIdx.x, lane = tid & 31, w = tid >> 5;
  int bid  = blockIdx.x;
  int cblk = bid % 24, tblk = (bid / 24) % 16, n = bid / (24 * 16);
  int t0 = tblk * 128, c0 = cblk * 64;
  int mrow = (w & 3) * 32, ncb = (w >> 2) * 32;
  int akk = tid >> 3, atb = (tid & 7) * 16;         // A: k-row (0..31), 16 t's
  int bnn = tid >> 2, bko = (tid & 3) * 8;          // B: n-row (0..63), 8 k's
  const float* abase = mask + ((size_t)n * EE + akk) * TT + t0 + atb;
  const unsigned short* bbase = w_inB + (size_t)(c0 + bnn) * EE + bko;

  float4 ar[4]; us8 br;
  auto load_tile = [&](int kt) {
    const float* ap = abase + (size_t)kt * 32 * TT;
    ar[0] = *(const float4*)(ap);
    ar[1] = *(const float4*)(ap + 4);
    ar[2] = *(const float4*)(ap + 8);
    ar[3] = *(const float4*)(ap + 12);
    br = *(const us8*)(bbase + kt * 32);
  };
  auto stage = [&](int buf) {
#pragma unroll
    for (int q = 0; q < 4; ++q) {
      float4 v = ar[q];
      As[buf][(atb + q * 4 + 0) * 40 + akk] = f2bf(v.x);
      As[buf][(atb + q * 4 + 1) * 40 + akk] = f2bf(v.y);
      As[buf][(atb + q * 4 + 2) * 40 + akk] = f2bf(v.z);
      As[buf][(atb + q * 4 + 3) * 40 + akk] = f2bf(v.w);
    }
    *(us8*)(&Bs[buf][bnn * 40 + bko]) = br;
  };

  load_tile(0);
  stage(0);
  __syncthreads();
  v8f acc00 = {}, acc01 = {}, acc10 = {}, acc11 = {};
  for (int kt = 0; kt < 16; ++kt) {
    int cur = kt & 1;
    if (kt + 1 < 16) load_tile(kt + 1);            // hide global latency behind WMMAs
    const unsigned short* Ab = &As[cur][0];
    const unsigned short* Bb = &Bs[cur][0];
    v16bf a0 = fragA(Ab, 40, mrow,      0, lane);
    v16bf a1 = fragA(Ab, 40, mrow + 16, 0, lane);
    v16bf b0 = fragBT(Bb, 40, ncb,      0, lane);
    v16bf b1 = fragBT(Bb, 40, ncb + 16, 0, lane);
    acc00 = wmma_bf16(a0, b0, acc00);
    acc01 = wmma_bf16(a0, b1, acc01);
    acc10 = wmma_bf16(a1, b0, acc10);
    acc11 = wmma_bf16(a1, b1, acc11);
    if (kt + 1 < 16) stage(1 - cur);
    __syncthreads();
  }

  int nloc = lane & 15, hi = (lane & 16) ? 8 : 0;
  v8f accs[2][2] = {{acc00, acc01}, {acc10, acc11}};
#pragma unroll
  for (int am = 0; am < 2; ++am)
#pragma unroll
    for (int bn = 0; bn < 2; ++bn) {
      v8f acc = accs[am][bn];
      int cg = c0 + ncb + bn * 16 + nloc;
      float bias = b_in[cg];
      float scale = (cg < EE) ? 0.125f : 1.0f;     // fold 1/sqrt(64) into q
#pragma unroll
      for (int r = 0; r < 8; ++r) {
        int tg = t0 + mrow + am * 16 + r + hi;
        qkv[((size_t)(n * TT + tg)) * E3 + cg] = f2bf((acc[r] + bias) * scale);
      }
    }
}

// ---------------- kernel 3: banded attention, one wave per (n, h, 16-query tile) ----------------
__global__ __launch_bounds__(32) void attn_kernel(const unsigned short* __restrict__ qkv,
                                                  unsigned short* __restrict__ obf) {
  __shared__ unsigned short Qs[16 * 64];    // [q][d]
  __shared__ unsigned short Ks[16 * 64];    // [key][d]  == BT layout for scores
  __shared__ unsigned short Vt[64 * 160];   // [d][key]  == BT layout for P@V
  __shared__ unsigned short Ps[16 * 160];   // [q][key]
  int lane = threadIdx.x;
  int bid = blockIdx.x;
  int qt = bid % 128, h = (bid / 128) % HH, n = bid / (128 * HH);
  int q0 = qt * 16;
  int kb0 = q0 - 64;

  for (int i = lane; i < 64 * 16; i += 32) Vt[(i >> 4) * 160 + 144 + (i & 15)] = 0;
  for (int i = lane; i < 16 * 16; i += 32) Ps[(i >> 4) * 160 + 144 + (i & 15)] = 0;

  const unsigned short* base = qkv + ((size_t)n * TT) * E3 + h * DD;
  const unsigned short* qbase = base + (size_t)q0 * E3;
#pragma unroll
  for (int i = 0; i < 4; ++i) {
    int off = (i * 32 + lane) * 8;
    int r = off >> 6, c = off & 63;
    *(us8*)(&Qs[r * 64 + c]) = *(const us8*)(qbase + (size_t)r * E3 + c);
  }
  __syncthreads();
  v16bf aq0 = fragA(Qs, 64, 0, 0, lane);
  v16bf aq1 = fragA(Qs, 64, 0, 32, lane);

  v8f s[9];
  for (int g = 0; g < 9; ++g) {
    int kbase = kb0 + g * 16;
#pragma unroll
    for (int i = 0; i < 4; ++i) {
      int off = (i * 32 + lane) * 8;
      int j = off >> 6, d = off & 63;
      int key = kbase + j;
      int kc = key < 0 ? 0 : (key > TT - 1 ? TT - 1 : key);
      const unsigned short* kv = base + (size_t)kc * E3;
      *(us8*)(&Ks[j * 64 + d]) = *(const us8*)(kv + EE + d);
      us8 vv = *(const us8*)(kv + 2 * EE + d);
#pragma unroll
      for (int e = 0; e < 8; ++e)
        Vt[(d + e) * 160 + g * 16 + j] = vv[e];
    }
    __syncthreads();
    v8f c = {};
    c = wmma_bf16(aq0, fragBT(Ks, 64, 0, 0, lane), c);
    c = wmma_bf16(aq1, fragBT(Ks, 64, 0, 32, lane), c);
    s[g] = c;
    __syncthreads();
  }

  int nloc = lane & 15, hi = (lane & 16) ? 8 : 0;
  float rmax[8], rsum[8];
#pragma unroll
  for (int r = 0; r < 8; ++r) rmax[r] = -3.0e38f;
  for (int g = 0; g < 9; ++g) {
    int key = kb0 + g * 16 + nloc;
    bool kval = (key >= 0) && (key < TT);
#pragma unroll
    for (int r = 0; r < 8; ++r) {
      int dq = (q0 + r + hi) - key;
      bool ok = kval && dq <= 64 && dq >= -64;
      float v = ok ? s[g][r] : -3.0e38f;
      s[g][r] = v;
      rmax[r] = fmaxf(rmax[r], v);
    }
  }
#pragma unroll
  for (int r = 0; r < 8; ++r) {
    float m = rmax[r];
    for (int off = 8; off; off >>= 1) m = fmaxf(m, __shfl_xor(m, off, 16));
    rmax[r] = m;
    rsum[r] = 0.0f;
  }
  for (int g = 0; g < 9; ++g)
#pragma unroll
    for (int r = 0; r < 8; ++r) {
      float p = __expf(s[g][r] - rmax[r]);
      s[g][r] = p;
      rsum[r] += p;
    }
#pragma unroll
  for (int r = 0; r < 8; ++r) {
    float t = rsum[r];
    for (int off = 8; off; off >>= 1) t += __shfl_xor(t, off, 16);
    rsum[r] = 1.0f / t;
  }
  for (int g = 0; g < 9; ++g)
#pragma unroll
    for (int r = 0; r < 8; ++r)
      Ps[(r + hi) * 160 + g * 16 + nloc] = f2bf(s[g][r] * rsum[r]);
  __syncthreads();

  v8f o0 = {}, o1 = {}, o2 = {}, o3 = {};
  for (int c = 0; c < 5; ++c) {
    v16bf a = fragA(Ps, 160, 0, c * 32, lane);
    o0 = wmma_bf16(a, fragBT(Vt, 160, 0,  c * 32, lane), o0);
    o1 = wmma_bf16(a, fragBT(Vt, 160, 16, c * 32, lane), o1);
    o2 = wmma_bf16(a, fragBT(Vt, 160, 32, c * 32, lane), o2);
    o3 = wmma_bf16(a, fragBT(Vt, 160, 48, c * 32, lane), o3);
  }
  unsigned short* ob = obf + ((size_t)(n * TT + q0)) * EE + h * DD;
#pragma unroll
  for (int r = 0; r < 8; ++r) {
    size_t row = (size_t)(r + hi) * EE;
    ob[row + 0  + nloc] = f2bf(o0[r]);
    ob[row + 16 + nloc] = f2bf(o1[r]);
    ob[row + 32 + nloc] = f2bf(o2[r]);
    ob[row + 48 + nloc] = f2bf(o3[r]);
  }
}

// ---------------- kernels 4/5: bf16 A @ B(BT) + bias, fused epilogue; 128x64 tile, double-buffered.
// mode 0: y = relu(A@B + b_out) * encodings  -> ybf [n][t][e] bf16
// mode 1: out = A@B + b_ct                   -> f32 [n][e][t] (transposed store)
__global__ __launch_bounds__(256) void proj_kernel(const unsigned short* __restrict__ A,
                                                   const unsigned short* __restrict__ B,
                                                   const float* __restrict__ bias,
                                                   const float* __restrict__ enc,
                                                   unsigned short* __restrict__ ybf,
                                                   float* __restrict__ outp, int mode) {
  __shared__ unsigned short As[2][128 * 40];
  __shared__ unsigned short Bs[2][64 * 40];
  int tid = threadIdx.x, lane = tid & 31, w = tid >> 5;
  int bid  = blockIdx.x;
  int cblk = bid % 8, tblk = (bid / 8) % 16, n = bid / (8 * 16);
  int t0 = tblk * 128, c0 = cblk * 64;
  int mrow = (w & 3) * 32, ncb = (w >> 2) * 32;
  int arr = tid >> 1, ako = (tid & 1) * 16;        // A: row (0..127), 16 k's
  int bnn = tid >> 2, bko = (tid & 3) * 8;         // B: row (0..63), 8 k's
  const unsigned short* abase = A + ((size_t)(n * TT + t0 + arr)) * EE + ako;
  const unsigned short* bbase = B + (size_t)(c0 + bnn) * EE + bko;

  us8 ar0, ar1, br;
  auto load_tile = [&](int kt) {
    const unsigned short* ap = abase + kt * 32;
    ar0 = *(const us8*)(ap);
    ar1 = *(const us8*)(ap + 8);
    br  = *(const us8*)(bbase + kt * 32);
  };
  auto stage = [&](int buf) {
    *(us8*)(&As[buf][arr * 40 + ako])     = ar0;
    *(us8*)(&As[buf][arr * 40 + ako + 8]) = ar1;
    *(us8*)(&Bs[buf][bnn * 40 + bko])     = br;
  };

  load_tile(0);
  stage(0);
  __syncthreads();
  v8f acc00 = {}, acc01 = {}, acc10 = {}, acc11 = {};
  for (int kt = 0; kt < 16; ++kt) {
    int cur = kt & 1;
    if (kt + 1 < 16) load_tile(kt + 1);
    const unsigned short* Ab = &As[cur][0];
    const unsigned short* Bb = &Bs[cur][0];
    v16bf a0 = fragA(Ab, 40, mrow,      0, lane);
    v16bf a1 = fragA(Ab, 40, mrow + 16, 0, lane);
    v16bf b0 = fragBT(Bb, 40, ncb,      0, lane);
    v16bf b1 = fragBT(Bb, 40, ncb + 16, 0, lane);
    acc00 = wmma_bf16(a0, b0, acc00);
    acc01 = wmma_bf16(a0, b1, acc01);
    acc10 = wmma_bf16(a1, b0, acc10);
    acc11 = wmma_bf16(a1, b1, acc11);
    if (kt + 1 < 16) stage(1 - cur);
    __syncthreads();
  }

  int nloc = lane & 15, hi = (lane & 16) ? 8 : 0;
  v8f accs[2][2] = {{acc00, acc01}, {acc10, acc11}};
#pragma unroll
  for (int am = 0; am < 2; ++am)
#pragma unroll
    for (int bn = 0; bn < 2; ++bn) {
      v8f acc = accs[am][bn];
      int cg = c0 + ncb + bn * 16 + nloc;
      float bv = bias[cg];
#pragma unroll
      for (int r = 0; r < 8; ++r) {
        int tg = t0 + mrow + am * 16 + r + hi;
        float v = acc[r] + bv;
        if (mode == 0) {
          v = fmaxf(v, 0.0f) * enc[((size_t)n * EE + cg) * TT + tg];
          ybf[((size_t)(n * TT + tg)) * EE + cg] = f2bf(v);
        } else {
          outp[((size_t)(n * EE + cg)) * TT + tg] = v;
        }
      }
    }
}

extern "C" void kernel_launch(void* const* d_in, const int* in_sizes, int n_in,
                              void* d_out, int out_size, void* d_ws, size_t ws_size,
                              hipStream_t stream) {
  const float* mask  = (const float*)d_in[0];
  const float* enc   = (const float*)d_in[1];
  const float* w_in  = (const float*)d_in[2];
  const float* b_in  = (const float*)d_in[3];
  const float* w_out = (const float*)d_in[4];
  const float* b_out = (const float*)d_in[5];
  const float* w_ct  = (const float*)d_in[6];
  const float* b_ct  = (const float*)d_in[7];
  float* out = (float*)d_out;

  char* ws = (char*)d_ws;
  size_t off = 0;
  auto carve = [&](size_t bytes) -> void* {
    void* p = ws + off;
    off = (off + bytes + 255) & ~(size_t)255;
    return p;
  };
  unsigned short* qkv    = (unsigned short*)carve((size_t)NB * TT * E3 * 2);
  unsigned short* obf    = (unsigned short*)carve((size_t)NB * TT * EE * 2);
  unsigned short* ybf    = (unsigned short*)carve((size_t)NB * TT * EE * 2);
  unsigned short* w_inB  = (unsigned short*)carve((size_t)E3 * EE * 2);
  unsigned short* w_outB = (unsigned short*)carve((size_t)EE * EE * 2);
  unsigned short* w_ctB  = (unsigned short*)carve((size_t)EE * EE * 2);

  convert_weights<<<dim3(512), dim3(256), 0, stream>>>(w_in, w_out, w_ct, w_inB, w_outB, w_ctB);
  qkv_kernel<<<dim3(NB * 16 * 24), dim3(256), 0, stream>>>(mask, w_inB, b_in, qkv);
  attn_kernel<<<dim3(NB * HH * 128), dim3(32), 0, stream>>>(qkv, obf);
  proj_kernel<<<dim3(NB * 16 * 8), dim3(256), 0, stream>>>(obf, w_outB, b_out, enc, ybf, nullptr, 0);
  proj_kernel<<<dim3(NB * 16 * 8), dim3(256), 0, stream>>>(ybf, w_ctB, b_ct, nullptr, nullptr, out, 1);
}
